// DeformableConv_13950053778309
// MI455X (gfx1250) — matrix operand order repaired and checked by reading, not compiled, not run on hardware
//
#include <hip/hip_runtime.h>
#include <hip/hip_bf16.h>
#include <math.h>

// Problem constants
#define NB    16
#define CIN   128
#define COUT  128
#define KK    3
#define PADX  1
#define PP    9
#define HH    64
#define WW    64
#define HWSZ  4096           // 64*64
#define NPIX  (NB*HWSZ)      // 65536 output pixels

typedef __attribute__((ext_vector_type(16))) _Float16 v16h;
typedef __attribute__((ext_vector_type(8)))  float    v8f;
typedef __attribute__((ext_vector_type(4)))  float    v4f;

// ---------------- Kernel 1: NCHW -> NHWC transpose (f32) -------------------
__global__ void k_transpose(const float* __restrict__ x, float* __restrict__ xt) {
    int pix = blockIdx.x;           // n*HW + y*W + xw
    int c   = threadIdx.x;          // 0..127
    int n   = pix >> 12;
    int hw  = pix & (HWSZ - 1);
    xt[(size_t)pix * CIN + c] = x[((size_t)(n * CIN + c)) * HWSZ + hw];
}

// ---------------- Kernel 2: offset conv (scalar f32, precise) --------------
__global__ void k_offset_conv(const float* __restrict__ xt,
                              const float* __restrict__ ow,
                              const float* __restrict__ ob,
                              float* __restrict__ off) {
    int idx = blockIdx.x * blockDim.x + threadIdx.x;   // (n*18+ch)*4096 + hw
    if (idx >= NB * 18 * HWSZ) return;
    int hw = idx & (HWSZ - 1);
    int t  = idx >> 12;
    int ch = t % 18;
    int n  = t / 18;
    int ho = hw >> 6, wo = hw & 63;
    float acc = ob[ch];
    for (int ky = 0; ky < 3; ++ky) {
        int yy = ho + ky - PADX;
        if ((unsigned)yy >= (unsigned)HH) continue;
        for (int kx = 0; kx < 3; ++kx) {
            int xx = wo + kx - PADX;
            if ((unsigned)xx >= (unsigned)WW) continue;
            const float* xv = xt + ((size_t)((n * HH + yy) * WW + xx)) * CIN;
            const float* wv = ow + (size_t)ch * CIN * PP + ky * 3 + kx;
            float s = 0.f;
            #pragma unroll 4
            for (int c = 0; c < CIN; ++c) s += xv[c] * wv[c * PP];
            acc += s;
        }
    }
    off[idx] = acc;
}

// ------- Kernel 3: pack deform_w into per-lane B-fragment layout (f16) -----
// wp[(((p*4+cc)*8+t)*32+lane)*16+h] = w[cout = t*16 + lane%16][cin = cc*32 + 16*(lane/16) + h][p]
__global__ void k_pack_w(const float* __restrict__ dw, _Float16* __restrict__ wp) {
    int idx = blockIdx.x * blockDim.x + threadIdx.x;   // 9*4*8*32*16 = 147456
    if (idx >= PP * 4 * 8 * 32 * 16) return;
    int h  = idx & 15;
    int L  = (idx >> 4) & 31;
    int tc = (idx >> 9) & 7;
    int cc = (idx >> 12) & 3;
    int p  = idx >> 14;
    int g  = L >> 4;
    int cout = tc * 16 + (L & 15);
    int cin  = cc * 32 + g * 16 + h;
    wp[idx] = (_Float16)dw[((size_t)cout * CIN + cin) * PP + p];
}

// ---------------- Kernel 4: deformable implicit GEMM via WMMA --------------
// Block: 128 threads (4 waves). Each block: 64 output pixels x all 128 couts.
// Each wave: 16 pixels (M) x 8 cout tiles of 16 (N). K = CIN*P = 1152, f16 WMMA.
__global__ void __launch_bounds__(128)
k_deform_main(const float* __restrict__ xt, const float* __restrict__ off,
              const _Float16* __restrict__ wp,
              const float* __restrict__ gamma, const float* __restrict__ beta,
              const float* __restrict__ rmean, const float* __restrict__ rvar,
              float* __restrict__ out) {
    __shared__ __align__(16) _Float16 sA[64 * CIN];   // 16 KB staged A tile
    __shared__ int   sY0[64], sX0[64];
    __shared__ float sFy[64], sFx[64];

    const int tid  = threadIdx.x;
    const int wave = tid >> 5;
    const int lane = tid & 31;
    const int g    = lane >> 4;       // half-wave group
    const int mn   = lane & 15;       // row (A) / column (B,C,D) index

    const int pixBase = blockIdx.x * 64;          // 4096 % 64 == 0 -> single n
    const int n       = pixBase >> 12;
    const int hwBase  = pixBase & (HWSZ - 1);

    v8f acc[8];
    v8f zer = {0.f, 0.f, 0.f, 0.f, 0.f, 0.f, 0.f, 0.f};
    #pragma unroll
    for (int t = 0; t < 8; ++t) acc[t] = zer;

    for (int p = 0; p < PP; ++p) {
        // ---- per-pixel sampling metadata for this tap ----
        if (tid < 64) {
            int hw = hwBase + tid;
            int ho = hw >> 6, wo = hw & 63;
            float dy = off[(size_t)(n * 18 + 2 * p) * HWSZ + hw];
            float dx = off[(size_t)(n * 18 + 2 * p + 1) * HWSZ + hw];
            float py = dy + (float)(p / 3) + (float)(ho - PADX);
            float px = dx + (float)(p % 3) + (float)(wo - PADX);
            float fy0 = floorf(py), fx0 = floorf(px);
            sY0[tid] = (int)fy0;  sX0[tid] = (int)fx0;
            sFy[tid] = py - fy0;  sFx[tid] = px - fx0;
        }
        __syncthreads();

        // ---- stage bilinear-sampled A tile into LDS (f16) ----
        {
            int pixel = tid >> 1;
            int cbase = (tid & 1) * 64;
            int y0 = sY0[pixel], x0 = sX0[pixel];
            float fy = sFy[pixel], fx = sFx[pixel];
            int y1 = y0 + 1, x1 = x0 + 1;
            bool vy0 = (unsigned)y0 < (unsigned)HH, vy1 = (unsigned)y1 < (unsigned)HH;
            bool vx0 = (unsigned)x0 < (unsigned)WW, vx1 = (unsigned)x1 < (unsigned)WW;
            int yc0 = min(max(y0, 0), HH - 1), yc1 = min(max(y1, 0), HH - 1);
            int xc0 = min(max(x0, 0), WW - 1), xc1 = min(max(x1, 0), WW - 1);
            float w00 = (1.f - fy) * (1.f - fx) * ((vy0 && vx0) ? 1.f : 0.f);
            float w01 = (1.f - fy) * fx         * ((vy0 && vx1) ? 1.f : 0.f);
            float w10 = fy * (1.f - fx)         * ((vy1 && vx0) ? 1.f : 0.f);
            float w11 = fy * fx                 * ((vy1 && vx1) ? 1.f : 0.f);
            const v4f* p00 = (const v4f*)(xt + ((size_t)((n * HH + yc0) * WW + xc0)) * CIN + cbase);
            const v4f* p01 = (const v4f*)(xt + ((size_t)((n * HH + yc0) * WW + xc1)) * CIN + cbase);
            const v4f* p10 = (const v4f*)(xt + ((size_t)((n * HH + yc1) * WW + xc0)) * CIN + cbase);
            const v4f* p11 = (const v4f*)(xt + ((size_t)((n * HH + yc1) * WW + xc1)) * CIN + cbase);
            _Float16* dst = &sA[pixel * CIN + cbase];
            #pragma unroll
            for (int c = 0; c < 16; ++c) {               // 16 x float4 = 64 channels
                v4f a = p00[c], b = p01[c], cv = p10[c], d = p11[c];
                #pragma unroll
                for (int j = 0; j < 4; ++j) {
                    float v = w00 * a[j] + w01 * b[j] + w10 * cv[j] + w11 * d[j];
                    dst[c * 4 + j] = (_Float16)v;
                }
            }
        }
        __syncthreads();

        // ---- K loop over this tap's 128 channels: 4 chunks of K=32 ----
        #pragma unroll
        for (int cc = 0; cc < 4; ++cc) {
            // A fragment: contiguous K = 16*g + h (matches packed B)
            int row = wave * 16 + mn;
            union { v4f f[2]; v16h h; } ua;
            ua.f[0] = *(const v4f*)&sA[row * CIN + cc * 32 + g * 16];
            ua.f[1] = *(const v4f*)&sA[row * CIN + cc * 32 + g * 16 + 8];
            #pragma unroll
            for (int t = 0; t < 8; ++t) {
                // B fragment: pre-packed, 32B contiguous per lane
                const v4f* bt = (const v4f*)(wp + ((size_t)(((p * 4 + cc) * 8 + t) * 32 + lane)) * 16);
                union { v4f f[2]; v16h h; } ub;
                ub.f[0] = bt[0]; ub.f[1] = bt[1];
                acc[t] = __builtin_amdgcn_wmma_f32_16x16x32_f16(
                    false, ua.h, false, ub.h, (short)0, acc[t], false, false);
            }
        }
        __syncthreads();
    }

    // ---- epilogue: BatchNorm + SiLU, direct store ----
    #pragma unroll
    for (int t = 0; t < 8; ++t) {
        int cout = t * 16 + mn;
        float sc = gamma[cout] * rsqrtf(rvar[cout] + 1e-5f);
        float sh = beta[cout] - rmean[cout] * sc;
        #pragma unroll
        for (int r = 0; r < 8; ++r) {
            int M   = r + 8 * g;                       // C/D layout: VGPR r, half-wave g
            int pix = pixBase + wave * 16 + M;
            int hw  = pix & (HWSZ - 1);
            float v = acc[t][r] * sc + sh;
            float s = v / (1.f + __expf(-v));          // v * sigmoid(v)
            out[((size_t)(n * COUT + cout)) * HWSZ + hw] = s;
        }
    }
}

// ---------------------------- launcher -------------------------------------
extern "C" void kernel_launch(void* const* d_in, const int* in_sizes, int n_in,
                              void* d_out, int out_size, void* d_ws, size_t ws_size,
                              hipStream_t stream) {
    const float* x    = (const float*)d_in[0];   // (16,128,64,64)
    const float* ow   = (const float*)d_in[1];   // (18,128,3,3)
    const float* ob   = (const float*)d_in[2];   // (18,)
    const float* dw   = (const float*)d_in[3];   // (128,128,3,3)
    const float* gam  = (const float*)d_in[4];
    const float* bet  = (const float*)d_in[5];
    const float* rme  = (const float*)d_in[6];
    const float* rva  = (const float*)d_in[7];
    float* out = (float*)d_out;

    // Workspace layout
    const size_t XT_BYTES  = (size_t)NB * HWSZ * CIN * sizeof(float);   // 33.55 MB
    const size_t OFF_BYTES = (size_t)NB * 18 * HWSZ * sizeof(float);    //  4.72 MB
    char* ws = (char*)d_ws;
    float*     xt  = (float*)ws;
    float*     off = (float*)(ws + XT_BYTES);
    _Float16*  wpk = (_Float16*)(ws + XT_BYTES + OFF_BYTES);            // 294 KB

    // 1) NCHW -> NHWC
    k_transpose<<<NPIX, CIN, 0, stream>>>(x, xt);
    // 2) offset conv
    {
        int total = NB * 18 * HWSZ;
        k_offset_conv<<<(total + 255) / 256, 256, 0, stream>>>(xt, ow, ob, off);
    }
    // 3) pack weights into B-fragment layout
    {
        int total = PP * 4 * 8 * 32 * 16;
        k_pack_w<<<(total + 255) / 256, 256, 0, stream>>>(dw, wpk);
    }
    // 4) main deformable-conv implicit GEMM (WMMA) + BN + SiLU
    k_deform_main<<<NPIX / 64, 128, 0, stream>>>(xt, off, wpk, gam, bet, rme, rva, out);
}
